// GNN3_2946347565064
// MI455X (gfx1250) — matrix operand order repaired and checked by reading, not compiled, not run on hardware
//
#include <hip/hip_runtime.h>
#include <hip/hip_bf16.h>
#include <math.h>

#define D_IN  128
#define D_OUT 256

typedef __attribute__((ext_vector_type(16))) __bf16 v16bf;
typedef __attribute__((ext_vector_type(8)))  __bf16 v8bf;
typedef __attribute__((ext_vector_type(8)))  float  v8f;

#define SHUF16(lo, hi) __builtin_shufflevector((lo), (hi), 0,1,2,3,4,5,6,7,8,9,10,11,12,13,14,15)

__device__ __forceinline__ float lrelu(float x, float s) { return x > 0.f ? x : s * x; }

// float atomic max via integer atomics (works with -inf init)
__device__ __forceinline__ void atomicMaxF(float* addr, float val) {
  if (val >= 0.f) atomicMax((int*)addr, __float_as_int(val));
  else            atomicMin((unsigned int*)addr, __float_as_uint(val));
}

// ---------------------------------------------------------------- init
__global__ void init_ws_k(float* amax, float* denom, float* agg, float* sumsq,
                          int n, int total) {
  int t = blockIdx.x * blockDim.x + threadIdx.x;
  if (t < total) agg[t] = 0.f;
  if (t < n) { amax[t] = -__builtin_inff(); denom[t] = 0.f; }
  if (t == 0) *sumsq = 0.f;
}

// ---------------------------------------------------------------- converts
__global__ void cvt_bf16_k(const float* __restrict__ in, __hip_bfloat16* __restrict__ out, int total) {
  int t = blockIdx.x * blockDim.x + threadIdx.x;
  if (t < total) out[t] = __float2bfloat16(in[t]);
}

// W [D_IN x D_OUT] f32 row-major  ->  Wt [D_OUT x D_IN] bf16 row-major
__global__ void transpose_w_k(const float* __restrict__ W, __hip_bfloat16* __restrict__ Wt) {
  int t = blockIdx.x * blockDim.x + threadIdx.x;
  if (t >= D_IN * D_OUT) return;
  int n = t >> 7;          // 0..255
  int k = t & (D_IN - 1);  // 0..127
  Wt[t] = __float2bfloat16(W[k * D_OUT + n]);
}

// ---------------------------------------------------------------- WMMA GEMM
// C[M x N] f32 = A[M x K] bf16 row-major  x  Bt[N x K] bf16 row-major (i.e. B = Bt^T)
// Block: 128 threads = 4 waves (2x2), wave tile 64x64, block tile 128x128, K-step 32.
template <bool SIG>
__global__ __launch_bounds__(128) void gemm_bf16_nt(
    const __bf16* __restrict__ A, const __bf16* __restrict__ Bt,
    float* __restrict__ C, int M, int N, int K) {
  const int lane = threadIdx.x & 31;
  const int wave = threadIdx.x >> 5;
  const int wm = wave >> 1, wn = wave & 1;
  const int rowBase = blockIdx.y * 128 + wm * 64;
  const int colBase = blockIdx.x * 128 + wn * 64;
  const int lm = lane & 15;   // row (A) / col (B) within 16-tile
  const int lh = lane >> 4;   // half-wave selector

  v8f acc[4][4];
#pragma unroll
  for (int i = 0; i < 4; i++)
#pragma unroll
    for (int j = 0; j < 4; j++) acc[i][j] = (v8f){};

  for (int kb = 0; kb < K; kb += 32) {
    v16bf af[4], bfr[4];
#pragma unroll
    for (int mi = 0; mi < 4; mi++) {
      int r = rowBase + mi * 16 + lm;
      r = r < M ? r : M - 1;
      // A layout: lane<16 -> K 0..7 (V0-3), 16..23 (V4-7); lane>=16 -> K 8..15, 24..31
      const __bf16* p = A + (size_t)r * K + kb + lh * 8;
      v8bf lo = *(const v8bf*)p;
      v8bf hi = *(const v8bf*)(p + 16);
      af[mi] = SHUF16(lo, hi);
    }
#pragma unroll
    for (int ni = 0; ni < 4; ni++) {
      int c = colBase + ni * 16 + lm;
      c = c < N ? c : N - 1;
      // B layout: lane<16 -> K 0..7, 16..23; lane>=16 -> K 8..15, 24..31 (same as A)
      const __bf16* p = Bt + (size_t)c * K + kb + lh * 8;
      v8bf lo = *(const v8bf*)p;
      v8bf hi = *(const v8bf*)(p + 16);
      bfr[ni] = SHUF16(lo, hi);
    }
    if (kb + 32 < K)
      __builtin_prefetch(A + (size_t)(rowBase + lm) * K + kb + 32, 0, 3);
#pragma unroll
    for (int mi = 0; mi < 4; mi++)
#pragma unroll
      for (int ni = 0; ni < 4; ni++)
        acc[mi][ni] = __builtin_amdgcn_wmma_f32_16x16x32_bf16(
            false, af[mi], false, bfr[ni], (short)0, acc[mi][ni], false, false);
  }

  // D layout: lane<16 -> N=lane, M = vgpr(0..7); lane>=16 -> N=lane-16, M = 8+vgpr
#pragma unroll
  for (int mi = 0; mi < 4; mi++) {
    int m0 = rowBase + mi * 16 + lh * 8;
#pragma unroll
    for (int ni = 0; ni < 4; ni++) {
      int n = colBase + ni * 16 + lm;
      if (n < N) {
#pragma unroll
        for (int v = 0; v < 8; v++) {
          int r = m0 + v;
          if (r < M) {
            float o = acc[mi][ni][v];
            if (SIG) o = 1.f / (1.f + __expf(-o));
            C[(size_t)r * N + n] = o;
          }
        }
      }
    }
  }
}

// ---------------------------------------------------------------- attention dots
__global__ void attn_dots_k(const float* __restrict__ h, const float* __restrict__ att_src,
                            const float* __restrict__ att_dst, float* __restrict__ a_src,
                            float* __restrict__ a_dst, int n) {
  int wid = (blockIdx.x * blockDim.x + threadIdx.x) >> 5;
  int lane = threadIdx.x & 31;
  if (wid >= n) return;
  const float* row = h + (size_t)wid * D_OUT;
  float s0 = 0.f, s1 = 0.f;
  for (int f = lane; f < D_OUT; f += 32) {
    float v = row[f];
    s0 += v * att_src[f];
    s1 += v * att_dst[f];
  }
  for (int off = 16; off > 0; off >>= 1) {
    s0 += __shfl_down(s0, off, 32);
    s1 += __shfl_down(s1, off, 32);
  }
  if (lane == 0) { a_src[wid] = s0; a_dst[wid] = s1; }
}

// ---------------------------------------------------------------- edge passes
__global__ void edge_alpha_k(const int* __restrict__ ei, const float* __restrict__ as,
                             const float* __restrict__ ad, float* __restrict__ alpha,
                             float* __restrict__ amax, int E, int ET) {
  int t = blockIdx.x * blockDim.x + threadIdx.x;
  if (t >= ET) return;
  int s, d;
  if (t < E) { s = ei[t]; d = ei[E + t]; } else { s = d = t - E; }
  float a = lrelu(as[s] + ad[d], 0.2f);
  alpha[t] = a;
  atomicMaxF(&amax[d], a);
}

__global__ void edge_exp_k(const int* __restrict__ ei, float* __restrict__ alpha,
                           const float* __restrict__ amax, float* __restrict__ denom,
                           int E, int ET) {
  int t = blockIdx.x * blockDim.x + threadIdx.x;
  if (t >= ET) return;
  int d = (t < E) ? ei[E + t] : t - E;
  float ex = __expf(alpha[t] - amax[d]);
  alpha[t] = ex;  // overwrite in place (fully rewritten each call)
  atomicAdd(&denom[d], ex);
}

// one wave per edge, 8 features per lane
__global__ void edge_scatter_k(const float* __restrict__ h, const float* __restrict__ ex,
                               const float* __restrict__ denom, const int* __restrict__ ei,
                               float* __restrict__ agg, int E, int ET) {
  int t = blockIdx.x * blockDim.x + threadIdx.x;
  int e = t >> 5;
  int lane = t & 31;
  if (e >= ET) return;
  int s, d;
  if (e < E) { s = ei[e]; d = ei[E + e]; } else { s = d = e - E; }
  float c = ex[e] / (denom[d] + 1e-16f);
  const float* hr = h + (size_t)s * D_OUT;
  float* ar = agg + (size_t)d * D_OUT;
#pragma unroll
  for (int i = 0; i < 8; i++) {
    int f = lane + i * 32;
    atomicAdd(&ar[f], hr[f] * c);
  }
}

// ---------------------------------------------------------------- bias+leaky+sumsq
__global__ void finalize_k(float* __restrict__ agg, const float* __restrict__ bias,
                           float* __restrict__ sumsq, int total) {
  int t = blockIdx.x * blockDim.x + threadIdx.x;
  float v = 0.f;
  if (t < total) {
    v = lrelu(agg[t] + bias[t & (D_OUT - 1)], 0.02f);
    agg[t] = v;
  }
  float s = v * v;
  for (int off = 16; off > 0; off >>= 1) s += __shfl_down(s, off, 32);
  __shared__ float ps[8];
  int lane = threadIdx.x & 31, w = threadIdx.x >> 5;
  if (lane == 0) ps[w] = s;
  __syncthreads();
  if (w == 0) {
    s = (lane < 8) ? ps[lane] : 0.f;
    for (int off = 4; off > 0; off >>= 1) s += __shfl_down(s, off, 32);
    if (lane == 0) atomicAdd(sumsq, s);
  }
}

__global__ void normalize_k(const float* __restrict__ agg, const float* __restrict__ sumsq,
                            __hip_bfloat16* __restrict__ hn, int total) {
  int t = blockIdx.x * blockDim.x + threadIdx.x;
  if (t >= total) return;
  float inv = rsqrtf(*sumsq);
  hn[t] = __float2bfloat16(agg[t] * inv);
}

// ---------------------------------------------------------------- host launch
extern "C" void kernel_launch(void* const* d_in, const int* in_sizes, int n_in,
                              void* d_out, int out_size, void* d_ws, size_t ws_size,
                              hipStream_t stream) {
  const float* x       = (const float*)d_in[0];
  const int*   ei      = (const int*)d_in[1];
  const float* W       = (const float*)d_in[2];
  const float* att_src = (const float*)d_in[3];
  const float* att_dst = (const float*)d_in[4];
  const float* bias    = (const float*)d_in[5];

  const int Nn = in_sizes[0] / D_IN;
  const int E  = in_sizes[1] / 2;
  const int ET = E + Nn;
  const int total = Nn * D_OUT;

  char* ws = (char*)d_ws;
  size_t off = 0;
  auto take = [&](size_t bytes) -> char* {
    char* p = ws + off;
    off += (bytes + 255) & ~(size_t)255;
    return p;
  };
  __hip_bfloat16* x_bf  = (__hip_bfloat16*)take((size_t)Nn * D_IN * 2);
  __hip_bfloat16* wt_bf = (__hip_bfloat16*)take((size_t)D_OUT * D_IN * 2);
  float* h     = (float*)take((size_t)Nn * D_OUT * 4);
  float* a_src = (float*)take((size_t)Nn * 4);
  float* a_dst = (float*)take((size_t)Nn * 4);
  float* alpha = (float*)take((size_t)ET * 4);
  float* amax  = (float*)take((size_t)Nn * 4);
  float* denom = (float*)take((size_t)Nn * 4);
  float* agg   = (float*)take((size_t)Nn * D_OUT * 4);
  __hip_bfloat16* hn = (__hip_bfloat16*)take((size_t)Nn * D_OUT * 2);
  float* sumsq = (float*)take(256);

  init_ws_k<<<(total + 255) / 256, 256, 0, stream>>>(amax, denom, agg, sumsq, Nn, total);
  cvt_bf16_k<<<((Nn * D_IN) + 255) / 256, 256, 0, stream>>>(x, x_bf, Nn * D_IN);
  transpose_w_k<<<((D_IN * D_OUT) + 255) / 256, 256, 0, stream>>>(W, wt_bf);

  dim3 g1((D_OUT + 127) / 128, (Nn + 127) / 128);
  gemm_bf16_nt<false><<<g1, 128, 0, stream>>>((const __bf16*)x_bf, (const __bf16*)wt_bf,
                                              h, Nn, D_OUT, D_IN);

  attn_dots_k<<<((Nn * 32) + 255) / 256, 256, 0, stream>>>(h, att_src, att_dst, a_src, a_dst, Nn);
  edge_alpha_k<<<(ET + 255) / 256, 256, 0, stream>>>(ei, a_src, a_dst, alpha, amax, E, ET);
  edge_exp_k<<<(ET + 255) / 256, 256, 0, stream>>>(ei, alpha, amax, denom, E, ET);
  edge_scatter_k<<<(int)(((size_t)ET * 32 + 255) / 256), 256, 0, stream>>>(h, alpha, denom, ei,
                                                                           agg, E, ET);
  finalize_k<<<(total + 255) / 256, 256, 0, stream>>>(agg, bias, sumsq, total);
  normalize_k<<<(total + 255) / 256, 256, 0, stream>>>(agg, sumsq, hn, total);

  dim3 g2((Nn + 127) / 128, (Nn + 127) / 128);
  gemm_bf16_nt<true><<<g2, 128, 0, stream>>>((const __bf16*)hn, (const __bf16*)hn,
                                             (float*)d_out, Nn, Nn, D_OUT);
}
